// CrystalDiffusionModel_72146860638420
// MI455X (gfx1250) — compile-verified, matrix operand order checked by reading, not run on hardware
//
#include <hip/hip_runtime.h>
#include <hip/hip_bf16.h>

#define N_NODES 100000
#define N_EDGES 1600000
#define HID 64
#define LAYERS 3

typedef __attribute__((ext_vector_type(16))) _Float16 v16h;
typedef __attribute__((ext_vector_type(8)))  _Float16 v8h;
typedef __attribute__((ext_vector_type(8)))  float    v8f;
typedef __attribute__((ext_vector_type(4)))  float    v4f;

union AF { v16h v; v8h h8[2]; _Float16 h[16]; };
union CF { v8f  v; float f[8]; };

__device__ __forceinline__ float siluf(float x) { return x / (1.f + __expf(-x)); }

// Load one 16x32 f16 WMMA A/B fragment slice for this lane:
// halves k0..k0+7 at p, halves k0+16..k0+23 at p+16 (caller folds k0 into p).
__device__ __forceinline__ v16h frag_ld(const _Float16* p) {
  AF u;
  u.h8[0] = *(const v8h*)(p);
  u.h8[1] = *(const v8h*)(p + 16);
  return u.v;
}

// Same, but converting from f32 source on the fly (node kernel: h/agg are f32).
__device__ __forceinline__ v16h frag_cvt(const float* p) {
  AF u;
  v4f a0 = *(const v4f*)(p);
  v4f a1 = *(const v4f*)(p + 4);
  v4f b0 = *(const v4f*)(p + 16);
  v4f b1 = *(const v4f*)(p + 20);
#pragma unroll
  for (int i = 0; i < 4; i++) {
    u.h[i]      = (_Float16)a0[i];
    u.h[4 + i]  = (_Float16)a1[i];
    u.h[8 + i]  = (_Float16)b0[i];
    u.h[12 + i] = (_Float16)b1[i];
  }
  return u.v;
}

__device__ __forceinline__ v8f wmma16(v16h a, v16h b, v8f c) {
  // 8 args: (neg_a, A, neg_b, B, c_mod, C, reuse_a, reuse_b)
  return __builtin_amdgcn_wmma_f32_16x16x32_f16(false, a, false, b, (short)0, c,
                                                false, false);
}

// ---------------------------------------------------------------------------
// Tiny kernels: time-embedding mean, weight transpose/cast, embed, zero, copy
// ---------------------------------------------------------------------------
__global__ void tmean_kernel(const float* __restrict__ t,
                             const float* __restrict__ tw1, const float* __restrict__ tb1,
                             const float* __restrict__ tw2, const float* __restrict__ tb2,
                             float* __restrict__ tmean) {
  __shared__ float sb[64];
  __shared__ float st[16];
  int n = threadIdx.x;  // 64 threads
  if (n < 16) st[n] = t[n];
  __syncthreads();
  float a = 0.f;
#pragma unroll
  for (int b = 0; b < 16; b++) a += siluf(st[b] * tw1[n] + tb1[n]);
  sb[n] = a * (1.f / 16.f);
  __syncthreads();
  float o = tb2[n];
  for (int k = 0; k < 64; k++) o += sb[k] * tw2[k * 64 + n];
  tmean[n] = o;
}

__global__ void prep_kernel(const float* __restrict__ ew1, const float* __restrict__ ew2,
                            const float* __restrict__ cw1, const float* __restrict__ nw1,
                            const float* __restrict__ nw2,
                            _Float16* __restrict__ w1T, float* __restrict__ w1d2,
                            _Float16* __restrict__ w2T, _Float16* __restrict__ c1T,
                            _Float16* __restrict__ nw1T, _Float16* __restrict__ nw2T) {
  int l = blockIdx.x;  // one block per layer
  const float* a = ew1 + (size_t)l * 129 * 64;
  _Float16* o1 = w1T + (size_t)l * 64 * 128;
  for (int i = threadIdx.x; i < 64 * 128; i += 256) {
    int n = i >> 7, k = i & 127;
    o1[i] = (_Float16)a[k * 64 + n];
  }
  for (int i = threadIdx.x; i < 64; i += 256) w1d2[l * 64 + i] = a[128 * 64 + i];

  const float* b1s = ew2 + (size_t)l * 64 * 64;
  _Float16* o2 = w2T + (size_t)l * 64 * 64;
  for (int i = threadIdx.x; i < 64 * 64; i += 256) {
    int n = i >> 6, k = i & 63;
    o2[i] = (_Float16)b1s[k * 64 + n];
  }
  const float* b2s = cw1 + (size_t)l * 64 * 64;
  _Float16* o3 = c1T + (size_t)l * 64 * 64;
  for (int i = threadIdx.x; i < 64 * 64; i += 256) {
    int n = i >> 6, k = i & 63;
    o3[i] = (_Float16)b2s[k * 64 + n];
  }
  const float* b3s = nw1 + (size_t)l * 128 * 64;
  _Float16* o4 = nw1T + (size_t)l * 64 * 128;
  for (int i = threadIdx.x; i < 64 * 128; i += 256) {
    int n = i >> 7, k = i & 127;
    o4[i] = (_Float16)b3s[k * 64 + n];
  }
  const float* b4s = nw2 + (size_t)l * 64 * 64;
  _Float16* o5 = nw2T + (size_t)l * 64 * 64;
  for (int i = threadIdx.x; i < 64 * 64; i += 256) {
    int n = i >> 6, k = i & 63;
    o5[i] = (_Float16)b4s[k * 64 + n];
  }
}

__global__ void embed_kernel(const int* __restrict__ z, const float* __restrict__ emb,
                             const float* __restrict__ tmean,
                             float* __restrict__ hf32, _Float16* __restrict__ h16) {
  int i = blockIdx.x * 256 + threadIdx.x;
  if (i >= N_NODES) return;
  const float* er = emb + (size_t)z[i] * 64;
  size_t o = (size_t)i * 64;
#pragma unroll 8
  for (int c = 0; c < 64; c++) {
    float v = er[c] + tmean[c];
    hf32[o + c] = v;
    h16[o + c] = (_Float16)v;
  }
}

__global__ void zero_kernel(float* __restrict__ p, long n) {
  long i = (long)blockIdx.x * 256 + threadIdx.x;
  if (i < n) p[i] = 0.f;
}

__global__ void copy_kernel(float* __restrict__ dst, const float* __restrict__ src, long n) {
  long i = (long)blockIdx.x * 256 + threadIdx.x;
  if (i < n) dst[i] = src[i];
}

// ---------------------------------------------------------------------------
// Edge kernel: per wave, 16 edges. Three chained WMMA GEMMs + atomics.
// ---------------------------------------------------------------------------
__global__ __launch_bounds__(256) void edge_kernel(
    const int* __restrict__ srcI, const int* __restrict__ dstI,
    const float* __restrict__ xcur, float* __restrict__ xnext,
    const _Float16* __restrict__ h16, float* __restrict__ agg,
    const _Float16* __restrict__ w1T, const float* __restrict__ w1d2,
    const _Float16* __restrict__ w2T, const _Float16* __restrict__ c1T,
    const float* __restrict__ eb1, const float* __restrict__ eb2,
    const float* __restrict__ cb1, const float* __restrict__ cw2,
    const float* __restrict__ cb2p) {
  __shared__ __align__(16) _Float16 sW1[64 * 128];
  __shared__ __align__(16) _Float16 sW2[64 * 64];
  __shared__ __align__(16) _Float16 sC1[64 * 64];
  __shared__ float sB1[64], sB2[64], sCB1[64], sCW2[64], sD2W[64];
  __shared__ __align__(16) _Float16 sStage[8][16 * 64];
  __shared__ float sD2[8][16];
  __shared__ float sDiff[8][3][16];
  __shared__ int sIdxD[8][16];
  __shared__ float sCw[8][16];

  int tid = threadIdx.x;
  {  // cooperative weight staging into LDS
    const int* g1 = (const int*)w1T;
    int* l1 = (int*)sW1;
    for (int i = tid; i < 64 * 128 / 2; i += 256) l1[i] = g1[i];
    const int* g2 = (const int*)w2T;
    int* l2 = (int*)sW2;
    for (int i = tid; i < 64 * 64 / 2; i += 256) l2[i] = g2[i];
    const int* g3 = (const int*)c1T;
    int* l3 = (int*)sC1;
    for (int i = tid; i < 64 * 64 / 2; i += 256) l3[i] = g3[i];
    if (tid < 64) {
      sB1[tid] = eb1[tid];
      sB2[tid] = eb2[tid];
      sCB1[tid] = cb1[tid];
      sCW2[tid] = cw2[tid];
      sD2W[tid] = w1d2[tid];
    }
  }
  __syncthreads();
  float c_b2 = cb2p[0];

  int wave = tid >> 5;
  int lane = tid & 31;
  int row = lane & 15;
  int hhi = lane >> 4;      // 0 / 1 half of the wave
  int moff = hhi * 8;       // C/D row offset for this half
  int koff = hhi * 8;       // A/B K offset for this half

  int e = blockIdx.x * 128 + wave * 16 + row;  // exact: E = 12500*128
  int ns = srcI[e];
  int nd = dstI[e];
  float dx = xcur[nd * 3 + 0] - xcur[ns * 3 + 0];
  float dy = xcur[nd * 3 + 1] - xcur[ns * 3 + 1];
  float dz = xcur[nd * 3 + 2] - xcur[ns * 3 + 2];
  float d2 = dx * dx + dy * dy + dz * dz;
  if (!hhi) {
    sD2[wave][row] = d2;
    sDiff[wave][0][row] = dx;
    sDiff[wave][1][row] = dy;
    sDiff[wave][2][row] = dz;
    sIdxD[wave][row] = nd;
  }
  // same-wave LDS ops are in-order on CDNA5: no barrier needed below.

  // ---- GEMM1: ef(16x129) @ e_w1 -> hidden(16x64); d2 column folded into C.
  CF acc[4];
#pragma unroll
  for (int nt = 0; nt < 4; nt++) {
    int n = nt * 16 + row;
    float b = sB1[n], wd = sD2W[n];
#pragma unroll
    for (int r = 0; r < 8; r++) acc[nt].f[r] = b + sD2[wave][r + moff] * wd;
  }
#pragma unroll
  for (int kc = 0; kc < 4; kc++) {
    const _Float16* hrow = h16 + (size_t)((kc < 2) ? nd : ns) * 64 + (kc & 1) * 32 + koff;
    v16h a = frag_ld(hrow);  // L2-resident gather
#pragma unroll
    for (int nt = 0; nt < 4; nt++) {
      v16h b = frag_ld(&sW1[(nt * 16 + row) * 128 + kc * 32 + koff]);
      acc[nt].v = wmma16(a, b, acc[nt].v);
    }
  }
  // silu -> f16 stage (C layout -> row-major for next A operand)
#pragma unroll
  for (int nt = 0; nt < 4; nt++) {
    int n = nt * 16 + row;
#pragma unroll
    for (int r = 0; r < 8; r++)
      sStage[wave][(r + moff) * 64 + n] = (_Float16)siluf(acc[nt].f[r]);
  }

  // ---- GEMM2: hidden @ e_w2 -> m(16x64)
  CF mac[4];
#pragma unroll
  for (int nt = 0; nt < 4; nt++) {
    float b = sB2[nt * 16 + row];
#pragma unroll
    for (int r = 0; r < 8; r++) mac[nt].f[r] = b;
  }
#pragma unroll
  for (int kc = 0; kc < 2; kc++) {
    v16h a = frag_ld(&sStage[wave][row * 64 + kc * 32 + koff]);
#pragma unroll
    for (int nt = 0; nt < 4; nt++) {
      v16h b = frag_ld(&sW2[(nt * 16 + row) * 64 + kc * 32 + koff]);
      mac[nt].v = wmma16(a, b, mac[nt].v);
    }
  }
#pragma unroll
  for (int nt = 0; nt < 4; nt++)
#pragma unroll
    for (int r = 0; r < 8; r++) mac[nt].f[r] = siluf(mac[nt].f[r]);
  // stage m for GEMM3 (loads above already issued; same-wave DS in-order)
#pragma unroll
  for (int nt = 0; nt < 4; nt++) {
    int n = nt * 16 + row;
#pragma unroll
    for (int r = 0; r < 8; r++)
      sStage[wave][(r + moff) * 64 + n] = (_Float16)mac[nt].f[r];
  }

  // ---- GEMM3: m @ c_w1 -> cwh(16x64)
  CF cac[4];
#pragma unroll
  for (int nt = 0; nt < 4; nt++) {
    float b = sCB1[nt * 16 + row];
#pragma unroll
    for (int r = 0; r < 8; r++) cac[nt].f[r] = b;
  }
#pragma unroll
  for (int kc = 0; kc < 2; kc++) {
    v16h a = frag_ld(&sStage[wave][row * 64 + kc * 32 + koff]);
#pragma unroll
    for (int nt = 0; nt < 4; nt++) {
      v16h b = frag_ld(&sC1[(nt * 16 + row) * 64 + kc * 32 + koff]);
      cac[nt].v = wmma16(a, b, cac[nt].v);
    }
  }
  // cw[m] = silu(cwh[m]) . c_w2 + c_b2  (reduce over the 16 lanes of each half)
  float part[8];
#pragma unroll
  for (int r = 0; r < 8; r++) {
    float s = 0.f;
#pragma unroll
    for (int nt = 0; nt < 4; nt++) s += siluf(cac[nt].f[r]) * sCW2[nt * 16 + row];
#pragma unroll
    for (int off = 8; off >= 1; off >>= 1) s += __shfl_xor(s, off, 16);
    part[r] = s;
  }
  if (row == 0) {
#pragma unroll
    for (int r = 0; r < 8; r++) sCw[wave][r + moff] = part[r] + c_b2;
  }
  // x update: x[dst] += diff * cw (lanes 0..15, one edge each)
  if (!hhi) {
    float cw = sCw[wave][row];
    int nn = sIdxD[wave][row];
    atomicAdd(&xnext[nn * 3 + 0], sDiff[wave][0][row] * cw);
    atomicAdd(&xnext[nn * 3 + 1], sDiff[wave][1][row] * cw);
    atomicAdd(&xnext[nn * 3 + 2], sDiff[wave][2][row] * cw);
  }
  // agg[dst] += m  (L2-resident atomics)
#pragma unroll
  for (int nt = 0; nt < 4; nt++) {
    int n = nt * 16 + row;
#pragma unroll
    for (int r = 0; r < 8; r++)
      atomicAdd(&agg[(size_t)sIdxD[wave][r + moff] * 64 + n], mac[nt].f[r]);
  }
}

// ---------------------------------------------------------------------------
// Node kernel: per wave, 16 nodes. h += silu([h|agg] @ n_w1 + b1) @ n_w2 + b2
// ---------------------------------------------------------------------------
__global__ __launch_bounds__(256) void node_kernel(
    float* __restrict__ hf32, _Float16* __restrict__ h16,
    const float* __restrict__ agg,
    const _Float16* __restrict__ w1T, const _Float16* __restrict__ w2T,
    const float* __restrict__ nb1, const float* __restrict__ nb2) {
  __shared__ __align__(16) _Float16 sW1[64 * 128];
  __shared__ __align__(16) _Float16 sW2[64 * 64];
  __shared__ float sB1[64], sB2[64];
  __shared__ __align__(16) _Float16 sStage[8][16 * 64];

  int tid = threadIdx.x;
  {
    const int* g1 = (const int*)w1T;
    int* l1 = (int*)sW1;
    for (int i = tid; i < 64 * 128 / 2; i += 256) l1[i] = g1[i];
    const int* g2 = (const int*)w2T;
    int* l2 = (int*)sW2;
    for (int i = tid; i < 64 * 64 / 2; i += 256) l2[i] = g2[i];
    if (tid < 64) {
      sB1[tid] = nb1[tid];
      sB2[tid] = nb2[tid];
    }
  }
  __syncthreads();

  int wave = tid >> 5;
  int lane = tid & 31;
  int row = lane & 15;
  int hhi = lane >> 4;
  int moff = hhi * 8;
  int koff = hhi * 8;

  int tile = blockIdx.x * 8 + wave;
  if (tile >= N_NODES / 16) return;  // uniform per wave; N divisible by 16
  int nodeBase = tile * 16;
  int node = nodeBase + row;

  // GEMM1: nf(16x128) @ n_w1 -> hidden
  CF acc[4];
#pragma unroll
  for (int nt = 0; nt < 4; nt++) {
    float b = sB1[nt * 16 + row];
#pragma unroll
    for (int r = 0; r < 8; r++) acc[nt].f[r] = b;
  }
#pragma unroll
  for (int kc = 0; kc < 4; kc++) {
    const float* src = (kc < 2) ? (hf32 + (size_t)node * 64 + (kc & 1) * 32 + koff)
                                : (agg + (size_t)node * 64 + (kc & 1) * 32 + koff);
    v16h a = frag_cvt(src);
#pragma unroll
    for (int nt = 0; nt < 4; nt++) {
      v16h b = frag_ld(&sW1[(nt * 16 + row) * 128 + kc * 32 + koff]);
      acc[nt].v = wmma16(a, b, acc[nt].v);
    }
  }
#pragma unroll
  for (int nt = 0; nt < 4; nt++) {
    int n = nt * 16 + row;
#pragma unroll
    for (int r = 0; r < 8; r++)
      sStage[wave][(r + moff) * 64 + n] = (_Float16)siluf(acc[nt].f[r]);
  }

  // GEMM2: hidden @ n_w2 -> out; residual writeback (f32 + f16 copies)
  CF oac[4];
#pragma unroll
  for (int nt = 0; nt < 4; nt++) {
    float b = sB2[nt * 16 + row];
#pragma unroll
    for (int r = 0; r < 8; r++) oac[nt].f[r] = b;
  }
#pragma unroll
  for (int kc = 0; kc < 2; kc++) {
    v16h a = frag_ld(&sStage[wave][row * 64 + kc * 32 + koff]);
#pragma unroll
    for (int nt = 0; nt < 4; nt++) {
      v16h b = frag_ld(&sW2[(nt * 16 + row) * 64 + kc * 32 + koff]);
      oac[nt].v = wmma16(a, b, oac[nt].v);
    }
  }
#pragma unroll
  for (int nt = 0; nt < 4; nt++) {
    int n = nt * 16 + row;
#pragma unroll
    for (int r = 0; r < 8; r++) {
      size_t idx = (size_t)(nodeBase + r + moff) * 64 + n;
      float hn = hf32[idx] + oac[nt].f[r];
      hf32[idx] = hn;
      h16[idx] = (_Float16)hn;
    }
  }
}

// ---------------------------------------------------------------------------
extern "C" void kernel_launch(void* const* d_in, const int* in_sizes, int n_in,
                              void* d_out, int out_size, void* d_ws, size_t ws_size,
                              hipStream_t stream) {
  (void)in_sizes; (void)n_in; (void)out_size; (void)ws_size;
  const float* x = (const float*)d_in[0];
  const int* z = (const int*)d_in[1];
  const float* t = (const float*)d_in[2];
  const int* eidx = (const int*)d_in[3];
  const float* emb = (const float*)d_in[4];
  const float* tw1 = (const float*)d_in[5];
  const float* tb1 = (const float*)d_in[6];
  const float* tw2 = (const float*)d_in[7];
  const float* tb2 = (const float*)d_in[8];
  const float* ew1 = (const float*)d_in[9];
  const float* eb1 = (const float*)d_in[10];
  const float* ew2 = (const float*)d_in[11];
  const float* eb2 = (const float*)d_in[12];
  const float* cw1 = (const float*)d_in[13];
  const float* cb1 = (const float*)d_in[14];
  const float* cw2 = (const float*)d_in[15];
  const float* cb2 = (const float*)d_in[16];
  const float* nw1 = (const float*)d_in[17];
  const float* nb1 = (const float*)d_in[18];
  const float* nw2 = (const float*)d_in[19];
  const float* nb2 = (const float*)d_in[20];
  float* out = (float*)d_out;

  char* ws = (char*)d_ws;
  size_t off = 0;
  auto alloc = [&](size_t b) -> char* {
    char* p = ws + off;
    off += (b + 255) & ~(size_t)255;
    return p;
  };
  float*     tmean = (float*)alloc(64 * 4);
  _Float16*  w1T   = (_Float16*)alloc((size_t)LAYERS * 64 * 128 * 2);
  float*     w1d2  = (float*)alloc((size_t)LAYERS * 64 * 4);
  _Float16*  w2T   = (_Float16*)alloc((size_t)LAYERS * 64 * 64 * 2);
  _Float16*  c1T   = (_Float16*)alloc((size_t)LAYERS * 64 * 64 * 2);
  _Float16*  nw1T  = (_Float16*)alloc((size_t)LAYERS * 64 * 128 * 2);
  _Float16*  nw2T  = (_Float16*)alloc((size_t)LAYERS * 64 * 64 * 2);
  float*     hf32  = (float*)alloc((size_t)N_NODES * 64 * 4);
  _Float16*  h16   = (_Float16*)alloc((size_t)N_NODES * 64 * 2);
  float*     agg   = (float*)alloc((size_t)N_NODES * 64 * 4);
  float*     xA    = (float*)alloc((size_t)N_NODES * 3 * 4);
  float*     xB    = (float*)alloc((size_t)N_NODES * 3 * 4);

  const int* srcI = eidx;
  const int* dstI = eidx + N_EDGES;

  tmean_kernel<<<1, 64, 0, stream>>>(t, tw1, tb1, tw2, tb2, tmean);
  prep_kernel<<<LAYERS, 256, 0, stream>>>(ew1, ew2, cw1, nw1, nw2, w1T, w1d2, w2T,
                                          c1T, nw1T, nw2T);
  embed_kernel<<<(N_NODES + 255) / 256, 256, 0, stream>>>(z, emb, tmean, hf32, h16);

  const float* xcur = x;
  for (int l = 0; l < LAYERS; l++) {
    float* xnext = (l == 0) ? xB : (l == 1) ? xA : out;
    long nx = (long)N_NODES * 3;
    copy_kernel<<<(int)((nx + 255) / 256), 256, 0, stream>>>(xnext, xcur, nx);
    long na = (long)N_NODES * 64;
    zero_kernel<<<(int)((na + 255) / 256), 256, 0, stream>>>(agg, na);
    edge_kernel<<<N_EDGES / 128, 256, 0, stream>>>(
        srcI, dstI, xcur, xnext, h16, agg, w1T + (size_t)l * 64 * 128, w1d2 + l * 64,
        w2T + (size_t)l * 64 * 64, c1T + (size_t)l * 64 * 64, eb1 + l * 64,
        eb2 + l * 64, cb1 + l * 64, cw2 + l * 64, cb2 + l);
    if (l < LAYERS - 1)  // final layer's h update is dead code in the reference
      node_kernel<<<(N_NODES / 16 + 7) / 8, 256, 0, stream>>>(
          hf32, h16, agg, nw1T + (size_t)l * 64 * 128, nw2T + (size_t)l * 64 * 64,
          nb1 + l * 64, nb2 + l * 64);
    xcur = xnext;
  }
}
